// MultiHeadAttention_3298534884011
// MI455X (gfx1250) — matrix operand order repaired and checked by reading, not compiled, not run on hardware
//
#include <hip/hip_runtime.h>

typedef __attribute__((__ext_vector_type__(16))) __bf16 v16bf;
typedef __attribute__((__ext_vector_type__(8)))  float  v8f;
typedef __attribute__((__ext_vector_type__(4)))  unsigned int u32x4;
typedef __attribute__((__ext_vector_type__(8)))  int i32x8;
typedef __attribute__((__ext_vector_type__(4)))  int i32x4;

constexpr int BATCH = 4;
constexpr int SEQ   = 2048;
constexpr int DIM   = 1024;
constexpr int NH    = 16;
constexpr int DK    = 64;

__device__ __forceinline__ v8f wmma_bf16(v16bf a, v16bf b, v8f c) {
  return __builtin_amdgcn_wmma_f32_16x16x32_bf16(
      false, a, false, b, (short)0, c, false, false);
}

// Low 32 bits of a generic pointer to LDS are the LDS byte offset (ISA 10.2).
__device__ __forceinline__ unsigned lds_off(const void* p) {
  return (unsigned)(uintptr_t)p;
}

// ---------------------------------------------------------------------------
// Tensor Data Mover: 2-D bf16 tile (tile_d0 elems x tile_d1 rows) from global
// (row stride = row_stride_elems) into LDS at lds_byte_off, with LDS row
// padding pad_amount dwords every pad_interval dwords (ISA ch.8 D# layout).
// data_size=1 (2 bytes). Descriptor groups 2/3 (and extra group) zero: 2-D.
// ---------------------------------------------------------------------------
__device__ __forceinline__ void tdm_load_2d(unsigned lds_byte_off,
                                            const void* gptr,
                                            unsigned tile_d0, unsigned tile_d1,
                                            unsigned row_stride_elems,
                                            unsigned pad_interval_code,
                                            unsigned pad_amount_code) {
  unsigned long long ga = (unsigned long long)(uintptr_t)gptr;
  u32x4 g0;
  g0[0] = 1u;                                        // count=1, user mode
  g0[1] = lds_byte_off;                              // lds_addr
  g0[2] = (unsigned)(ga & 0xFFFFFFFFull);            // global_addr[31:0]
  g0[3] = (unsigned)((ga >> 32) & 0x1FFFFFFull) | (2u << 30);  // [56:32], type=2
  i32x8 g1;
  g1[0] = (int)((1u << 16)                           // data_size = 2 bytes
              | (1u << 20)                           // pad_enable
              | (pad_interval_code << 22)
              | (pad_amount_code << 25));
  g1[1] = (int)((tile_d0 & 0xFFFFu) << 16);          // tensor_dim0 lo (63:48)
  g1[2] = (int)(((tile_d0 >> 16) & 0xFFFFu)          // tensor_dim0 hi (79:64)
              | ((tile_d1 & 0xFFFFu) << 16));        // tensor_dim1 lo (95:80)
  g1[3] = (int)(((tile_d1 >> 16) & 0xFFFFu)          // tensor_dim1 hi (111:96)
              | ((tile_d0 & 0xFFFFu) << 16));        // tile_dim0 (127:112)
  g1[4] = (int)(tile_d1 & 0xFFFFu);                  // tile_dim1; tile_dim2=0
  g1[5] = (int)row_stride_elems;                     // tensor_dim0_stride lo
  g1[6] = 0;
  g1[7] = 0;
  i32x4 z4 = {0, 0, 0, 0};
  i32x8 z8 = {0, 0, 0, 0, 0, 0, 0, 0};
  __builtin_amdgcn_tensor_load_to_lds(g0, g1, z4, z4, z8, 0);
}

// A fragment (16x32 bf16), ISA 7.12.2: lanes 0-15 M=lane K={0..7,16..23};
// lanes 16-31 M=lane-16 K={8..15,24..31}.
__device__ __forceinline__ v16bf lds_a_frag(const __bf16* base, int stride, int lane) {
  const int m  = lane & 15;
  const int kb = (lane < 16) ? 0 : 8;
  const __bf16* p = base + m * stride + kb;
  v16bf r;
#pragma unroll
  for (int e = 0; e < 8; ++e) r[e] = p[e];
#pragma unroll
  for (int e = 8; e < 16; ++e) r[e] = p[8 + e];
  return r;
}

// B fragment (32x16): lane = N column, K={0..7,16..23} (lanes 0-15) / {8..15,24..31} (lanes 16-31).
__device__ __forceinline__ v16bf lds_b_frag(const __bf16* base, int stride, int lane) {
  const int n  = lane & 15;
  const int kh = (lane < 16) ? 0 : 8;
  const __bf16* p = base + n * stride + kh;
  v16bf r;
#pragma unroll
  for (int e = 0; e < 8; ++e) r[e] = p[e];
#pragma unroll
  for (int e = 8; e < 16; ++e) r[e] = p[8 + e];
  return r;
}

// ---------------------------------------------------------------------------
// f32 -> bf16 conversion (weights): bandwidth-trivial pre-pass.
// ---------------------------------------------------------------------------
__global__ __launch_bounds__(256)
void cvt_f32_bf16_kernel(const float* __restrict__ x, __bf16* __restrict__ y, int n) {
  int i = (blockIdx.x * 256 + threadIdx.x) * 4;
  if (i < n) {
    float4 f = *(const float4*)(x + i);
    __bf16* d = y + i;
    d[0] = (__bf16)f.x; d[1] = (__bf16)f.y;
    d[2] = (__bf16)f.z; d[3] = (__bf16)f.w;
  }
}

// ---------------------------------------------------------------------------
// GEMM: Y[M,N] = A[M,K] * W[N,K]^T, block tile 128x128, BK=32, 128 threads
// (4 waves, 64x64 each => 16 WMMA / K-step / wave). Double-buffered LDS.
// W tiles always staged by TDM; A staged by TDM when bf16, else inline f32->bf16.
// ---------------------------------------------------------------------------
template <bool F32IN, bool SPLITOUT>
__global__ __launch_bounds__(128)
void gemm_bt_kernel(const void* __restrict__ Aptr,
                    const __bf16* __restrict__ Wb,
                    void* __restrict__ Yptr, int Kdim) {
  __shared__ __align__(16) __bf16 sA[2][128 * 40];   // 32 + 8 pad
  __shared__ __align__(16) __bf16 sB[2][128 * 40];

  const int tid  = threadIdx.x;
  const int lane = tid & 31;
  const int wave = tid >> 5;
  const int m0 = blockIdx.y * 128;
  const int n0 = blockIdx.x * 128;
  const int wm = (wave >> 1) * 64;
  const int wn = (wave & 1) * 64;

  v8f acc[4][4] = {};
  const int nk = Kdim / 32;

  auto stageA = [&](int kt, int buf) {
    if constexpr (F32IN) {
      const float* A = (const float*)Aptr;
#pragma unroll
      for (int i = 0; i < 8; ++i) {
        int id = tid + 128 * i;                  // 1024 float4 per tile
        int r = id >> 3, c = (id & 7) * 4;
        float4 f = *(const float4*)(A + (size_t)(m0 + r) * Kdim + kt * 32 + c);
        __bf16* d = sA[buf] + r * 40 + c;
        d[0] = (__bf16)f.x; d[1] = (__bf16)f.y;
        d[2] = (__bf16)f.z; d[3] = (__bf16)f.w;
      }
    } else {
      if (wave == 0)
        tdm_load_2d(lds_off(sA[buf]),
                    (const __bf16*)Aptr + (size_t)m0 * Kdim + kt * 32,
                    32, 128, Kdim, /*16 dw*/ 3, /*4 dw*/ 3);
    }
  };
  auto stageB = [&](int kt, int buf) {
    if (wave == 0)
      tdm_load_2d(lds_off(sB[buf]), Wb + (size_t)n0 * Kdim + kt * 32,
                  32, 128, Kdim, 3, 3);
  };

  stageA(0, 0);
  stageB(0, 0);
  if (wave == 0) __builtin_amdgcn_s_wait_tensorcnt(0);
  __syncthreads();

  for (int kt = 0; kt < nk; ++kt) {
    const int cur = kt & 1;
    if (kt + 1 < nk) { stageA(kt + 1, cur ^ 1); stageB(kt + 1, cur ^ 1); }

    v16bf afr[4], bfr[4];
#pragma unroll
    for (int mi = 0; mi < 4; ++mi)
      afr[mi] = lds_a_frag(sA[cur] + (wm + mi * 16) * 40, 40, lane);
#pragma unroll
    for (int ni = 0; ni < 4; ++ni)
      bfr[ni] = lds_b_frag(sB[cur] + (wn + ni * 16) * 40, 40, lane);
#pragma unroll
    for (int mi = 0; mi < 4; ++mi)
#pragma unroll
      for (int ni = 0; ni < 4; ++ni)
        acc[mi][ni] = wmma_bf16(afr[mi], bfr[ni], acc[mi][ni]);

    if (wave == 0) __builtin_amdgcn_s_wait_tensorcnt(0);
    __syncthreads();
  }

#pragma unroll
  for (int mi = 0; mi < 4; ++mi) {
#pragma unroll
    for (int ni = 0; ni < 4; ++ni) {
#pragma unroll
      for (int i = 0; i < 8; ++i) {
        int row = m0 + wm + mi * 16 + i + ((lane < 16) ? 0 : 8);
        int col = n0 + wn + ni * 16 + (lane & 15);
        float val = acc[mi][ni][i];
        if constexpr (SPLITOUT) {
          int b = row >> 11, s = row & (SEQ - 1);
          int h = col >> 6,  d = col & (DK - 1);
          ((__bf16*)Yptr)[(((size_t)b * NH + h) * SEQ + s) * DK + d] = (__bf16)val;
        } else {
          ((float*)Yptr)[(size_t)row * DIM + col] = val;
        }
      }
    }
  }
}

// ---------------------------------------------------------------------------
// Flash attention (causal). Block = 128 threads = 4 waves, 128 q rows/block,
// 32 q rows/wave. K tile staged by TDM, V staged transposed manually.
// ---------------------------------------------------------------------------
__global__ __launch_bounds__(128)
void flash_attn_kernel(const __bf16* __restrict__ Q,
                       const __bf16* __restrict__ Km,
                       const __bf16* __restrict__ Vm,
                       __bf16* __restrict__ Ctx) {
  __shared__ __align__(16) __bf16 sK[64 * 72];      // [key][dim], TDM-padded
  __shared__ __align__(16) __bf16 sV[64 * 72];      // transposed [dim][key]
  __shared__ __align__(16) __bf16 sP[4][32 * 72];   // per-wave P staging

  const int tid  = threadIdx.x;
  const int lane = tid & 31;
  const int wave = tid >> 5;
  const int b  = blockIdx.z;
  const int h  = blockIdx.y;
  const int qt = blockIdx.x;

  const size_t head_off = ((size_t)b * NH + h) * SEQ * DK;
  const __bf16* Qh = Q  + head_off;
  const __bf16* Kh = Km + head_off;
  const __bf16* Vh = Vm + head_off;

  const int q0 = qt * 128 + wave * 32;              // wave's first q row

  // Q fragments: 32 rows x 64 dims -> qa[mi][kf], loaded straight from global
  v16bf qa[2][2];
  {
    const int m  = lane & 15;
    const int kb = (lane < 16) ? 0 : 8;
#pragma unroll
    for (int mi = 0; mi < 2; ++mi) {
#pragma unroll
      for (int f = 0; f < 2; ++f) {
        const __bf16* p = Qh + (size_t)(q0 + mi * 16 + m) * DK + f * 32 + kb;
#pragma unroll
        for (int e = 0; e < 8; ++e)  qa[mi][f][e] = p[e];
#pragma unroll
        for (int e = 8; e < 16; ++e) qa[mi][f][e] = p[8 + e];
      }
    }
  }

  float mrun[2][8], lrun[2][8];
  v8f o[2][4] = {};
#pragma unroll
  for (int mi = 0; mi < 2; ++mi)
#pragma unroll
    for (int i = 0; i < 8; ++i) { mrun[mi][i] = -1e30f; lrun[mi][i] = 0.f; }

  const int nkt = 2 * qt + 2;                       // causal coverage of block
  for (int kt = 0; kt < nkt; ++kt) {
    __syncthreads();
    if (wave == 0)
      tdm_load_2d(lds_off(sK), Kh + (size_t)kt * 64 * DK,
                  64, 64, DK, /*32 dw*/ 4, /*4 dw*/ 3);
    // V transposed staging by all threads (TDM cannot transpose)
#pragma unroll
    for (int i = 0; i < 4; ++i) {
      int id = tid + 128 * i;                       // 512 x (8 bf16)
      int r = id >> 3, c = (id & 7) * 8;
      uint4 uv = *(const uint4*)(Vh + (size_t)(kt * 64 + r) * DK + c);
      const __bf16* e = (const __bf16*)&uv;
#pragma unroll
      for (int j = 0; j < 8; ++j) sV[(c + j) * 72 + r] = e[j];
    }
    if (wave == 0) __builtin_amdgcn_s_wait_tensorcnt(0);
    __syncthreads();

    // scores = Q K^T / sqrt(DK)
    v8f sc[2][4];
#pragma unroll
    for (int nf = 0; nf < 4; ++nf) {
      v16bf b0 = lds_b_frag(sK + (nf * 16) * 72 + 0,  72, lane);
      v16bf b1 = lds_b_frag(sK + (nf * 16) * 72 + 32, 72, lane);
#pragma unroll
      for (int mi = 0; mi < 2; ++mi) {
        v8f c = {};
        c = wmma_bf16(qa[mi][0], b0, c);
        c = wmma_bf16(qa[mi][1], b1, c);
        sc[mi][nf] = c * 0.125f;
      }
    }
    if (kt * 64 + 63 > q0) {                        // tile touches diagonal
#pragma unroll
      for (int mi = 0; mi < 2; ++mi)
#pragma unroll
        for (int nf = 0; nf < 4; ++nf)
#pragma unroll
          for (int i = 0; i < 8; ++i) {
            int row = q0 + mi * 16 + i + ((lane < 16) ? 0 : 8);
            int key = kt * 64 + nf * 16 + (lane & 15);
            if (key > row) sc[mi][nf][i] = -1e30f;
          }
    }

    // online softmax (rows live across 16 lanes of each half-wave)
#pragma unroll
    for (int mi = 0; mi < 2; ++mi) {
      float rm[8], alpha[8], rs[8];
#pragma unroll
      for (int i = 0; i < 8; ++i) {
        float v = fmaxf(fmaxf(sc[mi][0][i], sc[mi][1][i]),
                        fmaxf(sc[mi][2][i], sc[mi][3][i]));
#pragma unroll
        for (int off = 1; off < 16; off <<= 1)
          v = fmaxf(v, __shfl_xor(v, off, 32));
        rm[i] = v;
      }
#pragma unroll
      for (int i = 0; i < 8; ++i) {
        float mn = fmaxf(mrun[mi][i], rm[i]);
        alpha[i] = __expf(mrun[mi][i] - mn);
        mrun[mi][i] = mn;
        rs[i] = 0.f;
      }
#pragma unroll
      for (int nf = 0; nf < 4; ++nf)
#pragma unroll
        for (int i = 0; i < 8; ++i) {
          float p = __expf(sc[mi][nf][i] - mrun[mi][i]);
          sc[mi][nf][i] = p;
          rs[i] += p;
        }
#pragma unroll
      for (int i = 0; i < 8; ++i) {
#pragma unroll
        for (int off = 1; off < 16; off <<= 1)
          rs[i] += __shfl_xor(rs[i], off, 32);
        lrun[mi][i] = lrun[mi][i] * alpha[i] + rs[i];
      }
#pragma unroll
      for (int nd = 0; nd < 4; ++nd)
#pragma unroll
        for (int i = 0; i < 8; ++i) o[mi][nd][i] *= alpha[i];
    }

    // C-layout f32 P -> A-layout bf16 via per-wave LDS staging
    __bf16* pw = sP[wave];
#pragma unroll
    for (int mi = 0; mi < 2; ++mi)
#pragma unroll
      for (int nf = 0; nf < 4; ++nf)
#pragma unroll
        for (int i = 0; i < 8; ++i) {
          int row = mi * 16 + i + ((lane < 16) ? 0 : 8);
          int col = nf * 16 + (lane & 15);
          pw[row * 72 + col] = (__bf16)sc[mi][nf][i];
        }
    v16bf pa[2][2];
#pragma unroll
    for (int mi = 0; mi < 2; ++mi) {
      pa[mi][0] = lds_a_frag(pw + (mi * 16) * 72 + 0,  72, lane);
      pa[mi][1] = lds_a_frag(pw + (mi * 16) * 72 + 32, 72, lane);
    }

    // O += P V
#pragma unroll
    for (int nd = 0; nd < 4; ++nd) {
      v16bf v0 = lds_b_frag(sV + (nd * 16) * 72 + 0,  72, lane);
      v16bf v1 = lds_b_frag(sV + (nd * 16) * 72 + 32, 72, lane);
#pragma unroll
      for (int mi = 0; mi < 2; ++mi) {
        o[mi][nd] = wmma_bf16(pa[mi][0], v0, o[mi][nd]);
        o[mi][nd] = wmma_bf16(pa[mi][1], v1, o[mi][nd]);
      }
    }
  }

  // normalize, write ctx [B,S,D] bf16 (merged heads)
#pragma unroll
  for (int mi = 0; mi < 2; ++mi)
#pragma unroll
    for (int nd = 0; nd < 4; ++nd)
#pragma unroll
      for (int i = 0; i < 8; ++i) {
        int row = q0 + mi * 16 + i + ((lane < 16) ? 0 : 8);
        int col = h * DK + nd * 16 + (lane & 15);
        float v = o[mi][nd][i] / lrun[mi][i];
        Ctx[((size_t)b * SEQ + row) * DIM + col] = (__bf16)v;
      }
}

// ---------------------------------------------------------------------------
extern "C" void kernel_launch(void* const* d_in, const int* in_sizes, int n_in,
                              void* d_out, int out_size, void* d_ws, size_t ws_size,
                              hipStream_t stream) {
  const float* q  = (const float*)d_in[0];
  const float* k  = (const float*)d_in[1];
  const float* v  = (const float*)d_in[2];
  const float* Wq = (const float*)d_in[3];
  const float* Wk = (const float*)d_in[4];
  const float* Wv = (const float*)d_in[5];
  const float* Wo = (const float*)d_in[6];
  // d_in[7] (mask) unused: causal structure applied analytically.
  float* out = (float*)d_out;

  const size_t wsz   = (size_t)DIM * DIM;           // 1,048,576
  const size_t elems = (size_t)BATCH * SEQ * DIM;   // 8,388,608
  __bf16* Wqb = (__bf16*)d_ws;
  __bf16* Wkb = Wqb + wsz;
  __bf16* Wvb = Wkb + wsz;
  __bf16* Wob = Wvb + wsz;
  __bf16* Qb  = Wob + wsz;                          // [B,H,S,DK]
  __bf16* Kb  = Qb + elems;
  __bf16* Vb  = Kb + elems;
  __bf16* Cx  = Vb + elems;                         // [B,S,D]

  // weights f32 -> bf16 (once)
  const int cvtBlocks = (int)(wsz / (4 * 256));
  cvt_f32_bf16_kernel<<<cvtBlocks, 256, 0, stream>>>(Wq, Wqb, (int)wsz);
  cvt_f32_bf16_kernel<<<cvtBlocks, 256, 0, stream>>>(Wk, Wkb, (int)wsz);
  cvt_f32_bf16_kernel<<<cvtBlocks, 256, 0, stream>>>(Wv, Wvb, (int)wsz);
  cvt_f32_bf16_kernel<<<cvtBlocks, 256, 0, stream>>>(Wo, Wob, (int)wsz);

  dim3 gG(DIM / 128, (BATCH * SEQ) / 128);          // (8, 64)
  gemm_bt_kernel<true, true><<<gG, 128, 0, stream>>>(q, Wqb, Qb, DIM);
  gemm_bt_kernel<true, true><<<gG, 128, 0, stream>>>(k, Wkb, Kb, DIM);
  gemm_bt_kernel<true, true><<<gG, 128, 0, stream>>>(v, Wvb, Vb, DIM);

  flash_attn_kernel<<<dim3(SEQ / 128, NH, BATCH), 128, 0, stream>>>(Qb, Kb, Vb, Cx);

  gemm_bt_kernel<false, false><<<gG, 128, 0, stream>>>(Cx, Wob, out, DIM);
}
